// MultiHeadAttention_35321811043105
// MI455X (gfx1250) — matrix operand order repaired and checked by reading, not compiled, not run on hardware
//
#include <hip/hip_runtime.h>

// ---------------------------------------------------------------------------
// MHA forward for gfx1250 (MI455X).  All matmuls via v_wmma_f32_16x16x32_bf16.
// B=2, S=2048, D=1024, H=16, d=64.  The reference's reshape splits the SEQ
// dim, so Qh[b,h] is the contiguous [2048 x 64] slab at (b*H+h)*S*d.
//
// Pipeline:
//   1) one-time fp32 -> bf16 conversion of activations + weights (workspace)
//   2) bf16 WMMA GEMMs for Q,K (row-major bf16 out) and V (slab-transposed
//      bf16 out so P@V fragments load contiguously)
//   3) per-(bh, 16-query-tile) wave32 attention: QK^T -> LDS fp32 strip,
//      softmax (wave32 shfl_xor halves), normalized probs -> 512MB attn out,
//      P@V with normalization folded into epilogue, O stored bf16
//   4) bf16 WMMA GEMM for the output projection (fp32 out)
// ---------------------------------------------------------------------------

typedef __attribute__((ext_vector_type(16))) __bf16 v16bf;
typedef __attribute__((ext_vector_type(8)))  __bf16 v8bf;
typedef __attribute__((ext_vector_type(8)))  float  v8f;

#define SEQ    2048
#define DMODEL 1024
#define DHEAD  64
#define NBH    32
#define SLAB   (SEQ * DHEAD)            // 131072 elems per (b,h) slab
#define PROJ_ELEMS (2u * 2048u * 1024u) // 4,194,304
#define LDS_STRIDE 2064                 // fp32 per score row in LDS (padded)

__device__ __forceinline__ v8f wmma_bf16(v16bf a, v16bf b, v8f c) {
  return __builtin_amdgcn_wmma_f32_16x16x32_bf16(
      false, a, false, b, (short)0, c, false, false);
}

__device__ __forceinline__ v16bf cat16(v8bf lo, v8bf hi) {
  return __builtin_shufflevector(lo, hi, 0, 1, 2, 3, 4, 5, 6, 7,
                                 8, 9, 10, 11, 12, 13, 14, 15);
}

__device__ __forceinline__ v8bf ldg8(const __bf16* p) {
  return *(const v8bf*)p;   // one global_load_b128
}

// ---------------------------------------------------------------------------
// fp32 -> bf16 elementwise (8 elems / thread)
// ---------------------------------------------------------------------------
__global__ __launch_bounds__(256)
void cvt_f32_bf16(const float* __restrict__ in, __bf16* __restrict__ out, int n) {
  const int i = (blockIdx.x * 256 + threadIdx.x) * 8;
  if (i >= n) return;
  v8bf o;
#pragma unroll
  for (int t = 0; t < 8; ++t) o[t] = (__bf16)in[i + t];
  *(v8bf*)(out + i) = o;
}

// ---------------------------------------------------------------------------
// GEMM: Out[M,N] = X[M,K]bf16 @ W[N,K]bf16^T + bias[N]f32
// MODE 0: fp32 row-major out   (final projection)
// MODE 1: bf16 row-major out   (Q, K, O-producer reuse)
// MODE 2: bf16 slab-transposed out for V:  Vt[bh][dd][k],
//         bh=m/128, k=(m%128)*16 + n/64, dd=n%64
// One wave: 16x64 tile (1 A-frag x 4 B-frags). Block=128 (4 waves).
// ---------------------------------------------------------------------------
template <int MODE>
__global__ __launch_bounds__(128)
void gemm_xwT_bias(const __bf16* __restrict__ X, const __bf16* __restrict__ W,
                   const float* __restrict__ bias, void* __restrict__ OutP,
                   int M, int N, int K) {
  const int lane  = threadIdx.x & 31;
  const int wave  = threadIdx.x >> 5;
  const int half  = lane >> 4;
  const int l16   = lane & 15;
  const int mbase = blockIdx.x * 16;
  const int nbase = (blockIdx.y * 4 + wave) * 64;

  v8f acc[4] = {};

  const __bf16* xrow  = X + (size_t)(mbase + l16) * K + half * 8;
  const __bf16* wbase = W + (size_t)(nbase + l16) * K + half * 8;

  for (int k0 = 0; k0 < K; k0 += 32) {
    // Issue all 10 loads of this k-step before any WMMA.
    v8bf a0 = ldg8(xrow + k0);
    v8bf a1 = ldg8(xrow + k0 + 16);
    v8bf b0[4], b1[4];
#pragma unroll
    for (int j = 0; j < 4; ++j) {
      const __bf16* wrow = wbase + (size_t)(j * 16) * K + k0;
      b0[j] = ldg8(wrow);
      b1[j] = ldg8(wrow + 16);
    }
    const v16bf afrag = cat16(a0, a1);
#pragma unroll
    for (int j = 0; j < 4; ++j)
      acc[j] = wmma_bf16(afrag, cat16(b0[j], b1[j]), acc[j]);
  }

  // C/D layout: VGPR i of lane -> row (i + 8*half), col l16.
#pragma unroll
  for (int j = 0; j < 4; ++j) {
    const int n = nbase + j * 16 + l16;
    const float b = bias[n];
#pragma unroll
    for (int i = 0; i < 8; ++i) {
      const int m = mbase + i + half * 8;
      const float v = acc[j][i] + b;
      if (MODE == 0) {
        ((float*)OutP)[(size_t)m * N + n] = v;
      } else if (MODE == 1) {
        ((__bf16*)OutP)[(size_t)m * N + n] = (__bf16)v;
      } else {  // MODE 2: V slab transpose
        const size_t idx = (size_t)(m >> 7) * SLAB +
                           (size_t)(n & 63) * SEQ +
                           (size_t)((m & 127) * 16 + (n >> 6));
        ((__bf16*)OutP)[idx] = (__bf16)v;
      }
    }
  }
}

// ---------------------------------------------------------------------------
// Attention: one wave32 workgroup per (bh, 16-query tile).
// LDS = 16 x LDS_STRIDE fp32 score strip (~132 KB dynamic, 2 WGs/WGP).
// ---------------------------------------------------------------------------
__global__ __launch_bounds__(32)
void attn_block_kernel(const __bf16* __restrict__ Q, const __bf16* __restrict__ Km,
                       const __bf16* __restrict__ Vt, float* __restrict__ attn_out,
                       __bf16* __restrict__ O) {
  extern __shared__ float sc[];
  const int lane = threadIdx.x;
  const int half = lane >> 4;
  const int l16  = lane & 15;
  const int qt   = blockIdx.x;           // 0..127
  const int bh   = blockIdx.y;           // 0..31
  const size_t blk = (size_t)bh * SLAB;
  const float scale = 0.125f;            // 1/sqrt(64)

  // ---- preload Q A-fragments (16x64 -> two 16x32) ----
  const __bf16* qrow = Q + blk + (size_t)(qt * 16 + l16) * DHEAD + half * 8;
  v16bf aq[2];
#pragma unroll
  for (int f = 0; f < 2; ++f)
    aq[f] = cat16(ldg8(qrow + f * 32), ldg8(qrow + f * 32 + 16));

  // ---- pass 1: score strip into LDS ----
  const __bf16* kbase = Km + blk + (size_t)l16 * DHEAD + half * 8;
  for (int kt = 0; kt < SEQ / 16; ++kt) {
    const __bf16* krow = kbase + (size_t)kt * 16 * DHEAD;
    v8bf b00 = ldg8(krow);
    v8bf b01 = ldg8(krow + 16);
    v8bf b10 = ldg8(krow + 32);
    v8bf b11 = ldg8(krow + 48);
    v8f acc = {};
    acc = wmma_bf16(aq[0], cat16(b00, b01), acc);
    acc = wmma_bf16(aq[1], cat16(b10, b11), acc);
    const int col = kt * 16 + l16;
#pragma unroll
    for (int i = 0; i < 8; ++i)
      sc[(i + half * 8) * LDS_STRIDE + col] = acc[i] * scale;
  }
  __syncthreads();

  // ---- softmax: lane handles half of row l16 (float4 LDS traffic) ----
  float* rowp = sc + l16 * LDS_STRIDE + half * (SEQ / 2);
  float4* rp4 = (float4*)rowp;
  float mx = -3.0e38f;
  for (int c = 0; c < SEQ / 8; ++c) {
    float4 v = rp4[c];
    mx = fmaxf(mx, fmaxf(fmaxf(v.x, v.y), fmaxf(v.z, v.w)));
  }
  mx = fmaxf(mx, __shfl_xor(mx, 16, 32));
  float sum = 0.f;
  for (int c = 0; c < SEQ / 8; ++c) {
    float4 v = rp4[c];
    v.x = __expf(v.x - mx); v.y = __expf(v.y - mx);
    v.z = __expf(v.z - mx); v.w = __expf(v.w - mx);
    rp4[c] = v;                          // keep UNnormalized exp in LDS
    sum += (v.x + v.y) + (v.z + v.w);
  }
  sum += __shfl_xor(sum, 16, 32);
  const float inv = 1.0f / sum;          // lanes r and r+16 hold row r's inv
  __syncthreads();

  // ---- normalized probabilities -> global attn output (coalesced f4) ----
  {
    float* abase = attn_out + ((size_t)bh * SEQ + (size_t)qt * 16) * SEQ;
    for (int rr = 0; rr < 16; ++rr) {
      const float invr = __shfl(inv, rr, 32);
      const float4* s4 = (const float4*)(sc + rr * LDS_STRIDE);
      float4* d4 = (float4*)(abase + (size_t)rr * SEQ);
      for (int c = lane; c < SEQ / 4; c += 32) {
        float4 v = s4[c];
        v.x *= invr; v.y *= invr; v.z *= invr; v.w *= invr;
        d4[c] = v;
      }
    }
  }

  // ---- pass 2: O = (P @ V) * inv ----
  v8f acc[4] = {};
  const __bf16* vtb = Vt + blk + (size_t)l16 * SEQ + half * 8;
  const float* prow = sc + l16 * LDS_STRIDE + half * 8;
  for (int ks = 0; ks < SEQ / 32; ++ks) {
    const int k0 = ks * 32;
    // B fragments first (global), 2 x b128 each
    v8bf vb0[4], vb1[4];
#pragma unroll
    for (int j = 0; j < 4; ++j) {
      const __bf16* vp = vtb + (size_t)(j * 16) * SEQ + k0;
      vb0[j] = ldg8(vp);
      vb1[j] = ldg8(vp + 16);
    }
    // A fragment from LDS exp values (fp32 -> bf16)
    v16bf afrag;
    const float* p0 = prow + k0;
#pragma unroll
    for (int t = 0; t < 8; ++t) {
      afrag[t]     = (__bf16)p0[t];
      afrag[8 + t] = (__bf16)p0[16 + t];
    }
#pragma unroll
    for (int j = 0; j < 4; ++j)
      acc[j] = wmma_bf16(afrag, cat16(vb0[j], vb1[j]), acc[j]);
  }

  __bf16* orow = O + blk + (size_t)(qt * 16) * DHEAD;
#pragma unroll
  for (int j = 0; j < 4; ++j) {
#pragma unroll
    for (int i = 0; i < 8; ++i) {
      const int m = i + half * 8;
      const float invm = __shfl(inv, m, 32);
      orow[(size_t)m * DHEAD + j * 16 + l16] = (__bf16)(acc[j][i] * invm);
    }
  }
}

// ---------------------------------------------------------------------------
// Workspace (bf16 elems):
//   Xq | Xk | Xv   : 3 x 4,194,304
//   Wq | Wk | Wv | Wo : 4 x 1,048,576
//   Qb | Kb | Vt | Ob : 4 x 4,194,304      total 32M bf16 = 64 MB
// ---------------------------------------------------------------------------
extern "C" void kernel_launch(void* const* d_in, const int* in_sizes, int n_in,
                              void* d_out, int out_size, void* d_ws, size_t ws_size,
                              hipStream_t stream) {
  const float* pre[3] = {(const float*)d_in[0], (const float*)d_in[1],
                         (const float*)d_in[2]};
  const float* Wf[4]  = {(const float*)d_in[3], (const float*)d_in[5],
                         (const float*)d_in[7], (const float*)d_in[9]};
  const float* bias[4] = {(const float*)d_in[4], (const float*)d_in[6],
                          (const float*)d_in[8], (const float*)d_in[10]};

  __bf16* wsb = (__bf16*)d_ws;
  __bf16* Xb[3];
  for (int i = 0; i < 3; ++i) Xb[i] = wsb + (size_t)i * PROJ_ELEMS;
  __bf16* Wb[4];
  for (int i = 0; i < 4; ++i)
    Wb[i] = wsb + 3 * (size_t)PROJ_ELEMS + (size_t)i * (DMODEL * DMODEL);
  __bf16* act = wsb + 3 * (size_t)PROJ_ELEMS + 4 * (size_t)(DMODEL * DMODEL);
  __bf16* Qb = act;
  __bf16* Kb = act + (size_t)PROJ_ELEMS;
  __bf16* Vt = act + 2 * (size_t)PROJ_ELEMS;
  __bf16* Ob = act + 3 * (size_t)PROJ_ELEMS;

  float* proj = (float*)d_out;
  float* attn = proj + PROJ_ELEMS;

  // 1) one-time conversions
  {
    const int nAct = (int)PROJ_ELEMS, nW = DMODEL * DMODEL;
    dim3 blk(256);
    dim3 gA((nAct / 8 + 255) / 256), gW((nW / 8 + 255) / 256);
    for (int i = 0; i < 3; ++i)
      cvt_f32_bf16<<<gA, blk, 0, stream>>>(pre[i], Xb[i], nAct);
    for (int i = 0; i < 4; ++i)
      cvt_f32_bf16<<<gW, blk, 0, stream>>>(Wf[i], Wb[i], nW);
  }

  const int M = 2 * SEQ, N = DMODEL, K = DMODEL;
  dim3 ggrid(M / 16, N / 256), gblk(128);

  // 2) projections
  gemm_xwT_bias<1><<<ggrid, gblk, 0, stream>>>(Xb[0], Wb[0], bias[0], Qb, M, N, K);
  gemm_xwT_bias<1><<<ggrid, gblk, 0, stream>>>(Xb[1], Wb[1], bias[1], Kb, M, N, K);
  gemm_xwT_bias<2><<<ggrid, gblk, 0, stream>>>(Xb[2], Wb[2], bias[2], Vt, M, N, K);

  // 3) attention
  dim3 agrid(SEQ / 16, NBH), ablk(32);
  size_t lds_bytes = (size_t)16 * LDS_STRIDE * sizeof(float);
  attn_block_kernel<<<agrid, ablk, lds_bytes, stream>>>(Qb, Kb, Vt, attn, Ob);

  // 4) output projection (fp32 out)
  gemm_xwT_bias<0><<<ggrid, gblk, 0, stream>>>(Ob, Wb[3], bias[3], proj, M, N, K);
}
